// Net_16647293239698
// MI455X (gfx1250) — matrix operand order repaired and checked by reading, not compiled, not run on hardware
//
#include <hip/hip_runtime.h>
#include <math.h>

// ---------------------------------------------------------------------------
// SplineCNN forward for gfx1250 (MI455X). Dense basis transforms + FC layers
// run on V_WMMA_F32_16X16X32_F16. The spline-layer GEMMs stage their shared
// 16xK A-tile into LDS with the Tensor Data Mover (tensor_load_to_lds +
// s_wait_tensorcnt), cutting A traffic 8x per block; waves then read A
// fragments via ds_load_b128 while streaming B from L2. XW intermediates are
// stored f16 (<=136MB -> resident in the 192MB global L2) so the per-edge
// corner gathers never go to HBM cold.
// ---------------------------------------------------------------------------

typedef __attribute__((ext_vector_type(16))) _Float16 v16h;
typedef __attribute__((ext_vector_type(8)))  _Float16 v8h;
typedef __attribute__((ext_vector_type(8)))  float    v8f;
typedef __attribute__((ext_vector_type(4)))  unsigned int u32x4;
typedef __attribute__((ext_vector_type(8)))  int      i32x8;
typedef __attribute__((ext_vector_type(4)))  int      i32x4;

#define TPB 256
#define CDIV(a, b) (((a) + (b) - 1) / (b))
#define ENC_NEG_INF 0x007FFFFFu  // ordered-uint encoding of -inf

// ------------------------------ elementwise --------------------------------
__global__ void fill_u32_kernel(unsigned* __restrict__ p, unsigned v, int n) {
  int i = blockIdx.x * blockDim.x + threadIdx.x;
  if (i < n) p[i] = v;
}

__global__ void zero_f32_kernel(float* __restrict__ p, int n) {
  int i = blockIdx.x * blockDim.x + threadIdx.x;
  if (i < n) p[i] = 0.0f;
}

// segment-max via monotonic uint encoding + atomicMax
__global__ void pool_scatter_kernel(const float* __restrict__ x,
                                    const int* __restrict__ cl,
                                    unsigned* __restrict__ keys,
                                    int total, int C) {
  int i = blockIdx.x * blockDim.x + threadIdx.x;
  if (i >= total) return;
  int n = i / C, c = i - n * C;
  unsigned u = __float_as_uint(x[i]);
  unsigned key = (u & 0x80000000u) ? ~u : (u | 0x80000000u);
  atomicMax(&keys[(size_t)cl[n] * C + c], key);
}

__global__ void pool_decode_kernel(const unsigned* __restrict__ keys,
                                   float* __restrict__ out, int n) {
  int i = blockIdx.x * blockDim.x + threadIdx.x;
  if (i >= n) return;
  unsigned key = keys[i];
  float f = 0.0f;                       // empty cell (-inf) -> 0
  if (key != ENC_NEG_INF) {
    unsigned u = (key & 0x80000000u) ? (key & 0x7FFFFFFFu) : ~key;
    f = __uint_as_float(u);
  }
  out[i] = f;
}

__global__ void to_half_pad_kernel(const float* __restrict__ in,
                                   _Float16* __restrict__ out,
                                   int total, int C, int Cpad) {
  int i = blockIdx.x * blockDim.x + threadIdx.x;
  if (i >= total) return;
  int n = i / Cpad, c = i - n * Cpad;
  out[i] = (_Float16)((c < C) ? in[(size_t)n * C + c] : 0.0f);
}

// Pack spline weights W[64,Ci,Co] (+ root Wr[Ci,Co] as kernel 64) into
// B-transposed f16 layout: out[(ker*Co+o) * Kp + i], K-pad with zeros.
__global__ void pack_spline_w_kernel(const float* __restrict__ W,
                                     const float* __restrict__ Wr,
                                     _Float16* __restrict__ out,
                                     int Ci, int Co, int Kp, int total) {
  int i = blockIdx.x * blockDim.x + threadIdx.x;
  if (i >= total) return;
  int row = i / Kp, k = i - row * Kp;
  int ker = row / Co, o = row - ker * Co;
  float v = 0.0f;
  if (k < Ci)
    v = (ker < 64) ? W[((size_t)ker * Ci + k) * Co + o] : Wr[(size_t)k * Co + o];
  out[i] = (_Float16)v;
}

// Pack FC weight [K, Nin] into B-transposed [Npad, K] f16 (zero-pad columns).
__global__ void pack_fc_kernel(const float* __restrict__ Win,
                               _Float16* __restrict__ out,
                               int K, int Nin, int total) {
  int i = blockIdx.x * blockDim.x + threadIdx.x;
  if (i >= total) return;
  int n = i / K, k = i - n * K;
  out[i] = (_Float16)((n < Nin) ? Win[(size_t)k * Nin + n] : 0.0f);
}

// ------------------------- WMMA fragment helpers ---------------------------
// A fragment (16x32 f16): lanes 0-15 row r hold K 0..15,
// lanes 16-31 row r hold K 16..31 (matches B order).
// B fragment (32x16 f16): lane = column (mod 16), lanes 0-15 hold K 0..15,
// lanes 16-31 hold K 16..31 -> one contiguous 32B read from B^T.
__device__ __forceinline__ v16h make_a_frag(const _Float16* arow, int k0) {
  v8h alo = *(const v8h*)(arow + k0);
  v8h ahi = *(const v8h*)(arow + k0 + 8);
  return __builtin_shufflevector(alo, ahi, 0, 1, 2, 3, 4, 5, 6, 7,
                                 8, 9, 10, 11, 12, 13, 14, 15);
}

// ----------------------- generic WMMA GEMM (global) ------------------------
// C[M,N] = A[M,K] * B[K,N], A row-major f16 (lda=K), Bt = B^T row-major f16
// (ldb=K), C f32 or f16 row-major (ldc=N). One wave per 16x16 tile, 8 waves
// per block. K multiple of 32, M,N multiples of 16. Used for the FC head.
__global__ void wmma_gemm_kernel(const _Float16* __restrict__ A,
                                 const _Float16* __restrict__ Bt,
                                 void* __restrict__ Cout, int half_out,
                                 int ntn, int total_tiles, int K,
                                 int lda, int ldb, int ldc) {
  int wave = threadIdx.x >> 5;
  int lane = threadIdx.x & 31;
  int tile = blockIdx.x * 8 + wave;
  if (tile >= total_tiles) return;          // wave-uniform: EXEC stays all-1s
  int tm = tile / ntn, tn = tile - tm * ntn;
  int r  = lane & 15;
  int hi = (lane >> 4) & 1;
  const _Float16* arow = A + (size_t)(tm * 16 + r) * lda + hi * 16;
  const _Float16* brow = Bt + (size_t)(tn * 16 + r) * ldb + hi * 16;
  v8f acc = {};
  for (int k0 = 0; k0 < K; k0 += 32) {
    v16h a = make_a_frag(arow, k0);
    v16h b = *(const v16h*)(brow + k0);
    acc = __builtin_amdgcn_wmma_f32_16x16x32_f16(false, a, false, b,
                                                 (short)0, acc, false, false);
  }
  int col = lane & 15;
  int rb  = hi * 8;
  if (half_out) {
    _Float16* C = (_Float16*)Cout;
#pragma unroll
    for (int v = 0; v < 8; ++v)
      C[(size_t)(tm * 16 + rb + v) * ldc + tn * 16 + col] = (_Float16)acc[v];
  } else {
    float* C = (float*)Cout;
#pragma unroll
    for (int v = 0; v < 8; ++v)
      C[(size_t)(tm * 16 + rb + v) * ldc + tn * 16 + col] = acc[v];
  }
}

// -------------------- spline GEMM: TDM-staged A in LDS ---------------------
// blockIdx.y = tm (shared by all 8 waves); the 16xK A tile (<=4KB) is staged
// into LDS once per block by the Tensor Data Mover, then each wave streams a
// different 16-column B strip from L2. K in {32,64,128}, multiple of 32.
// The LDS tile is a v8h array indexed directly (stays addrspace(3)) so
// fragment reads lower to ds_load_b128; an asm escape+memory clobber stops
// the compiler from proving As is never written (the TDM writes it).
__global__ void __launch_bounds__(256)
wmma_gemm_tdm_kernel(const _Float16* __restrict__ A,
                     const _Float16* __restrict__ Bt,
                     _Float16* __restrict__ C,
                     int ntn, int K, int ldc) {
  __shared__ v8h As[(16 * 128) / 8];  // sole LDS allocation -> lds offset 0
  const int tm = blockIdx.y;

#if __has_builtin(__builtin_amdgcn_tensor_load_to_lds)
  if (threadIdx.x < 32) {             // one wave issues the TDM descriptor
    unsigned long long ga =
        (unsigned long long)(const void*)(A + (size_t)tm * 16 * K);
    u32x4 g0;
    g0[0] = 1u;                                   // count=1 (valid user D#)
    g0[1] = 0u;                                   // lds_addr = 0
    g0[2] = (unsigned)(ga & 0xFFFFFFFFu);         // global_addr[31:0]
    g0[3] = (unsigned)((ga >> 32) & 0x01FFFFFFu)  // global_addr[56:32]
            | (2u << 30);                         // type = 2 ("image")
    i32x8 g1;
    g1[0] = 0x00010000;                 // data_size=2B; no pad/iter/multicast
    g1[1] = (K & 0xFFFF) << 16;         // tensor_dim0[15:0]
    g1[2] = (K >> 16) | (16 << 16);     // tensor_dim0[31:16] | tensor_dim1 lo
    g1[3] = (K << 16);                  // tensor_dim1 hi | tile_dim0 = K
    g1[4] = 16;                         // tile_dim1 = 16, tile_dim2 = 0
    g1[5] = K;                          // tensor_dim0_stride[31:0] = K
    g1[6] = 0;
    g1[7] = 0;
    i32x4 z4 = {0, 0, 0, 0};
#if __clang_major__ >= 23
    i32x8 z8 = {0, 0, 0, 0, 0, 0, 0, 0};
    __builtin_amdgcn_tensor_load_to_lds(g0, g1, z4, z4, z8, 0);
#else
    __builtin_amdgcn_tensor_load_to_lds(g0, g1, z4, z4, 0);
#endif
    __builtin_amdgcn_s_wait_tensorcnt(0);
  }
#else
  {
    const v8h* Ag = (const v8h*)(A + (size_t)tm * 16 * K);
    int nv = (16 * K) >> 3;
    for (int i = threadIdx.x; i < nv; i += 256) As[i] = Ag[i];
  }
#endif
  // Escape As + memory clobber: the TDM wrote LDS behind the compiler's
  // back, so prevent "never-stored global" load folding.
  {
    const void* asp = (const void*)As;
    asm volatile("" : "+v"(asp) : : "memory");
  }
  __syncthreads();

  const int wave = threadIdx.x >> 5;
  const int lane = threadIdx.x & 31;
  const int tn = blockIdx.x * 8 + wave;
  if (tn >= ntn) return;                 // wave-uniform (after the barrier)
  const int r  = lane & 15;
  const int hi = (lane >> 4) & 1;
  const int abase = (r * K + hi * 16) >> 3;  // in v8h units (K mult of 32)
  const _Float16* brow = Bt + (size_t)(tn * 16 + r) * K + hi * 16;
  v8f acc = {};
  for (int k0 = 0; k0 < K; k0 += 32) {
    v8h alo = As[abase + (k0 >> 3)];        // ds_load_b128
    v8h ahi = As[abase + (k0 >> 3) + 1];    // ds_load_b128
    v16h a = __builtin_shufflevector(alo, ahi, 0, 1, 2, 3, 4, 5, 6, 7,
                                     8, 9, 10, 11, 12, 13, 14, 15);
    v16h b = *(const v16h*)(brow + k0);
    acc = __builtin_amdgcn_wmma_f32_16x16x32_f16(false, a, false, b,
                                                 (short)0, acc, false, false);
  }
  const int col = lane & 15;
  const int rb  = hi * 8;
#pragma unroll
  for (int v = 0; v < 8; ++v)
    C[(size_t)(tm * 16 + rb + v) * ldc + tn * 16 + col] = (_Float16)acc[v];
}

// ----------------------------- spline edges --------------------------------
// One block per edge, one thread per output channel. Gathers the 8 corner
// rows of XW[src] (f16, L2-resident), blends with trilinear B-spline weights,
// atomically accumulates into agg[dst].
__global__ void spline_edges_kernel(const float* __restrict__ pseudo,
                                    const int* __restrict__ src,
                                    const int* __restrict__ dst,
                                    const _Float16* __restrict__ XW,
                                    float* __restrict__ agg,
                                    float* __restrict__ deg, int Co) {
  int e  = blockIdx.x;
  int co = threadIdx.x;
  float p0 = pseudo[e * 3 + 0] * 3.0f;
  float p1 = pseudo[e * 3 + 1] * 3.0f;
  float p2 = pseudo[e * 3 + 2] * 3.0f;
  float k0 = fminf(floorf(p0), 2.0f);
  float k1 = fminf(floorf(p1), 2.0f);
  float k2 = fminf(floorf(p2), 2.0f);
  float f0 = p0 - k0, f1 = p1 - k1, f2 = p2 - k2;
  int i0 = (int)k0, i1 = (int)k1, i2 = (int)k2;
  const _Float16* xw = XW + (size_t)src[e] * (65 * Co) + co;
  float m = 0.0f;
#pragma unroll
  for (int c = 0; c < 8; ++c) {
    int b0 = c & 1, b1 = (c >> 1) & 1, b2 = (c >> 2) & 1;
    float w = (b0 ? f0 : 1.0f - f0) * (b1 ? f1 : 1.0f - f1) *
              (b2 ? f2 : 1.0f - f2);
    int idx = (i0 + b0) + ((i1 + b1) << 2) + ((i2 + b2) << 4);
    m += w * (float)xw[(size_t)idx * Co];
  }
  atomicAdd(&agg[(size_t)dst[e] * Co + co], m);
  if (co == 0) atomicAdd(&deg[dst[e]], 1.0f);
}

// out = elu(agg/max(deg,1) + root_term + bias); root term is kernel #64 of XW
__global__ void spline_finalize_kernel(const float* __restrict__ agg,
                                       const float* __restrict__ deg,
                                       const _Float16* __restrict__ XW,
                                       const float* __restrict__ bias,
                                       float* __restrict__ out, int N, int Co) {
  int i = blockIdx.x * blockDim.x + threadIdx.x;
  if (i >= N * Co) return;
  int n = i / Co, co = i - n * Co;
  float d = deg[n];
  d = (d > 1.0f) ? d : 1.0f;
  float v = agg[i] / d + (float)XW[(size_t)n * (65 * Co) + 64 * Co + co] +
            bias[co];
  out[i] = (v > 0.0f) ? v : expm1f(v);
}

__global__ void fc1_post_kernel(const float* __restrict__ in,
                                const float* __restrict__ bias,
                                _Float16* __restrict__ out, int total, int C) {
  int i = blockIdx.x * blockDim.x + threadIdx.x;
  if (i >= total) return;
  int c = i % C;
  float v = in[i] + bias[c];
  out[i] = (_Float16)((v > 0.0f) ? v : expm1f(v));
}

// in: [32,16] f32 (cols 10..15 are zero-weight pad); out: [32,10] log-softmax
__global__ void fc2_logsoftmax_kernel(const float* __restrict__ in,
                                      const float* __restrict__ bias,
                                      float* __restrict__ out) {
  int r = threadIdx.x;  // 32 rows, one thread each
  float v[10], mx = -3.0e38f;
#pragma unroll
  for (int c = 0; c < 10; ++c) {
    v[c] = in[r * 16 + c] + bias[c];
    mx = fmaxf(mx, v[c]);
  }
  float s = 0.0f;
#pragma unroll
  for (int c = 0; c < 10; ++c) s += expf(v[c] - mx);
  float l = mx + logf(s);
#pragma unroll
  for (int c = 0; c < 10; ++c) out[r * 10 + c] = v[c] - l;
}

// ------------------------------- host side ---------------------------------
static inline size_t align256(size_t x) { return (x + 255) & ~(size_t)255; }

static void launch_gemm(const _Float16* A, const _Float16* Bt, void* C,
                        int half_out, int M, int N, int K, hipStream_t s) {
  int ntn = N / 16;
  int total_tiles = (M / 16) * ntn;
  int blocks = CDIV(total_tiles, 8);
  wmma_gemm_kernel<<<blocks, 256, 0, s>>>(A, Bt, C, half_out, ntn, total_tiles,
                                          K, K, K, N);
}

static void launch_gemm_tdm(const _Float16* A, const _Float16* Bt, _Float16* C,
                            int M, int N, int K, hipStream_t s) {
  int ntn = N / 16;
  dim3 grid(CDIV(ntn, 8), M / 16);
  wmma_gemm_tdm_kernel<<<grid, 256, 0, s>>>(A, Bt, C, ntn, K, N);
}

extern "C" void kernel_launch(void* const* d_in, const int* in_sizes, int n_in,
                              void* d_out, int out_size, void* d_ws,
                              size_t ws_size, hipStream_t stream) {
  (void)in_sizes; (void)n_in; (void)out_size; (void)ws_size;

  const float* x      = (const float*)d_in[0];
  const float* pseudo[4] = {(const float*)d_in[1], (const float*)d_in[2],
                            (const float*)d_in[3], (const float*)d_in[4]};
  const float* Ws[4]  = {(const float*)d_in[5],  (const float*)d_in[8],
                         (const float*)d_in[11], (const float*)d_in[14]};
  const float* Wrs[4] = {(const float*)d_in[6],  (const float*)d_in[9],
                         (const float*)d_in[12], (const float*)d_in[15]};
  const float* bs[4]  = {(const float*)d_in[7],  (const float*)d_in[10],
                         (const float*)d_in[13], (const float*)d_in[16]};
  const float* fc1_w = (const float*)d_in[17];
  const float* fc1_b = (const float*)d_in[18];
  const float* fc2_w = (const float*)d_in[19];
  const float* fc2_b = (const float*)d_in[20];
  const int* cluster0 = (const int*)d_in[21];
  const int* srcs[4] = {(const int*)d_in[22], (const int*)d_in[25],
                        (const int*)d_in[28], (const int*)d_in[31]};
  const int* dsts[4] = {(const int*)d_in[23], (const int*)d_in[26],
                        (const int*)d_in[29], (const int*)d_in[32]};
  const int* clus[4] = {(const int*)d_in[24], (const int*)d_in[27],
                        (const int*)d_in[30], (const int*)d_in[33]};

  // Layer geometry
  const int Nn[5]   = {32768, 16384, 8192, 4096, 256};  // nodes per stage
  const int CiA[4]  = {1, 32, 64, 128};
  const int KpA[4]  = {32, 32, 64, 128};                // K padded to >=32
  const int CoA[4]  = {32, 64, 128, 256};
  const int Me[4]   = {262144, 131072, 65536, 32768};

  // Workspace layout (~156 MB); XW in f16 keeps the largest intermediate
  // inside the 192MB global L2.
  char* ws = (char*)d_ws;
  size_t off = 0;
  _Float16* XW   = (_Float16*)(ws + off); off = align256(off + (size_t)32768 * 2080 * 2);
  unsigned* keys = (unsigned*)(ws + off); off = align256(off + (size_t)524288 * 4);
  float* pool_o  = (float*)(ws + off);    off = align256(off + (size_t)524288 * 4);
  float* h_elu   = (float*)(ws + off);    off = align256(off + (size_t)1048576 * 4);
  _Float16* feat = (_Float16*)(ws + off); off = align256(off + (size_t)1048576 * 2);
  float* agg     = (float*)(ws + off);    off = align256(off + (size_t)1048576 * 4);
  float* deg     = (float*)(ws + off);    off = align256(off + (size_t)32768 * 4);
  _Float16* wpk  = (_Float16*)(ws + off); off = align256(off + (size_t)2129920 * 2);
  float* fc2_out = (float*)(ws + off);    off = align256(off + (size_t)32 * 16 * 4);

  // ---- pool 0: x[65536,1] -> pool_o[32768,1] ----
  {
    int nk = 32768;
    fill_u32_kernel<<<CDIV(nk, TPB), TPB, 0, stream>>>(keys, ENC_NEG_INF, nk);
    pool_scatter_kernel<<<CDIV(65536, TPB), TPB, 0, stream>>>(x, cluster0, keys,
                                                              65536, 1);
    pool_decode_kernel<<<CDIV(nk, TPB), TPB, 0, stream>>>(keys, pool_o, nk);
  }

  // ---- spline conv layers ----
  for (int l = 0; l < 4; ++l) {
    int N = Nn[l], Ci = CiA[l], Kp = KpA[l], Co = CoA[l], M = Me[l];
    int Nnext = Nn[l + 1];

    int nfeat = N * Kp;
    to_half_pad_kernel<<<CDIV(nfeat, TPB), TPB, 0, stream>>>(pool_o, feat,
                                                             nfeat, Ci, Kp);
    int nw = 65 * Co * Kp;
    pack_spline_w_kernel<<<CDIV(nw, TPB), TPB, 0, stream>>>(Ws[l], Wrs[l], wpk,
                                                            Ci, Co, Kp, nw);
    // XW[N, 65*Co] = feat @ [W_0..W_63 | Wr]   (TDM-staged A, WMMA f16->f16)
    launch_gemm_tdm(feat, wpk, XW, N, 65 * Co, Kp, stream);

    zero_f32_kernel<<<CDIV(N * Co, TPB), TPB, 0, stream>>>(agg, N * Co);
    zero_f32_kernel<<<CDIV(N, TPB), TPB, 0, stream>>>(deg, N);

    spline_edges_kernel<<<M, Co, 0, stream>>>(pseudo[l], srcs[l], dsts[l], XW,
                                              agg, deg, Co);
    spline_finalize_kernel<<<CDIV(N * Co, TPB), TPB, 0, stream>>>(
        agg, deg, XW, bs[l], h_elu, N, Co);

    int nk = Nnext * Co;
    fill_u32_kernel<<<CDIV(nk, TPB), TPB, 0, stream>>>(keys, ENC_NEG_INF, nk);
    pool_scatter_kernel<<<CDIV(N * Co, TPB), TPB, 0, stream>>>(h_elu, clus[l],
                                                               keys, N * Co, Co);
    pool_decode_kernel<<<CDIV(nk, TPB), TPB, 0, stream>>>(keys, pool_o, nk);
  }

  // ---- FC head: pool_o is [256,256] == [32, 2048] row-major ----
  {
    int nfeat = 32 * 2048;
    to_half_pad_kernel<<<CDIV(nfeat, TPB), TPB, 0, stream>>>(pool_o, feat,
                                                             nfeat, 2048, 2048);
    int nw1 = 512 * 2048;
    pack_fc_kernel<<<CDIV(nw1, TPB), TPB, 0, stream>>>(fc1_w, wpk, 2048, 512,
                                                       nw1);
    launch_gemm(feat, wpk, pool_o, /*half_out=*/0, 32, 512, 2048, stream);

    fc1_post_kernel<<<CDIV(32 * 512, TPB), TPB, 0, stream>>>(pool_o, fc1_b,
                                                             feat, 32 * 512, 512);
    int nw2 = 16 * 512;
    pack_fc_kernel<<<CDIV(nw2, TPB), TPB, 0, stream>>>(fc2_w, wpk, 512, 10, nw2);
    launch_gemm(feat, wpk, fc2_out, /*half_out=*/0, 32, 16, 512, stream);

    fc2_logsoftmax_kernel<<<1, 32, 0, stream>>>(fc2_out, fc2_b, (float*)d_out);
  }
}
